// ProfileHMM_55439437857594
// MI455X (gfx1250) — matrix operand order repaired and checked by reading, not compiled, not run on hardware
//
#include <hip/hip_runtime.h>
#include <hip/hip_bf16.h>
#include <cstdint>

// ---------------- problem constants ----------------
#define Mlen   384
#define Aal    21
#define Bb     64
#define Ll     256
#define Kst    769          // 2M+1 states
#define KCOL   784          // 49*16 padded columns (targets)
#define KROW   800          // 25*32 padded rows (sources / WMMA K dim)
#define NT     49           // N tiles of 16
#define NKT    25           // K steps of 32
#define NEGL   (-1.0e32f)

typedef __attribute__((ext_vector_type(16))) _Float16 v16h;
typedef __attribute__((ext_vector_type(8)))  float    v8f;

union Frag { v16h h; float4 f[2]; };

// ---------------- workspace layout (bytes) ----------------
static constexpr size_t OFF_TPACK = 0;                                  // NT*NKT*1024 B (f16)
static constexpr size_t SZ_TPACK  = (size_t)NT * NKT * 1024;            // 1,254,400
static constexpr size_t OFF_OBS   = OFF_TPACK + SZ_TPACK;               // 21*784 f32
static constexpr size_t SZ_OBS    = (size_t)Aal * KCOL * 4;
static constexpr size_t OFF_INIT  = OFF_OBS + SZ_OBS;                   // 769 f32
static constexpr size_t SZ_INIT   = (size_t)Kst * 4;
static constexpr size_t OFF_LET   = OFF_INIT + SZ_INIT;                 // 64*256 i32
static constexpr size_t SZ_LET    = (size_t)Bb * Ll * 4;
static constexpr size_t OFF_SRCA  = OFF_LET + SZ_LET;                   // 769 f32 each
static constexpr size_t OFF_SRCB  = OFF_SRCA + Kst * 4;
static constexpr size_t OFF_ROWB  = OFF_SRCB + Kst * 4;
static constexpr size_t OFF_TT    = OFF_ROWB + Kst * 4;
static constexpr size_t OFF_C     = OFF_TT   + Kst * 4;                 // 386 f32
static constexpr size_t OFF_RN    = OFF_C    + 386 * 4;                 // 384*3*2 f32
static constexpr size_t OFF_UN    = OFF_RN   + (size_t)Mlen * 6 * 4;

// dynamic LDS layout for the scan kernel
static constexpr int LDS_ALPHA = 0;                        // 16*784 f32 = 50176
static constexpr int LDS_EA    = 50176;                    // NKT*1024  = 25600
static constexpr int LDS_MX    = 75776;                    // 16 f32
static constexpr int LDS_LT    = 75840;                    // 16 i32
static constexpr int LDS_BYTES = 75904;

__device__ __forceinline__ float lse2(float a, float b) {
  float m = fmaxf(a, b);
  return m + logf(expf(a - m) + expf(b - m));
}

// ---------------- kernel 1: setup small arrays ----------------
__global__ void hmm_setup_kernel(const float* __restrict__ pre,
                                 const float* __restrict__ iseq,
                                 const float* __restrict__ ins,
                                 const float* __restrict__ del,
                                 char* __restrict__ ws) {
  float* rn   = (float*)(ws + OFF_RN);
  float* un   = (float*)(ws + OFF_UN);
  float* C    = (float*)(ws + OFF_C);
  float* srcA = (float*)(ws + OFF_SRCA);
  float* srcB = (float*)(ws + OFF_SRCB);
  float* rowb = (float*)(ws + OFF_ROWB);
  float* tt   = (float*)(ws + OFF_TT);
  float* initv= (float*)(ws + OFF_INIT);
  float* obs  = (float*)(ws + OFF_OBS);
  const int tid = threadIdx.x;
  __shared__ float red[256];

  // log-softmax over the last (size-2) axis of insert/delete
  for (int idx = tid; idx < Mlen * 3; idx += 256) {
    float a = ins[idx * 2], b = ins[idx * 2 + 1];
    float l = lse2(a, b);
    rn[idx * 2] = a - l; rn[idx * 2 + 1] = b - l;
    a = del[idx * 2]; b = del[idx * 2 + 1];
    l = lse2(a, b);
    un[idx * 2] = a - l; un[idx * 2 + 1] = b - l;
  }
  __syncthreads();

  // delete-chain cumsum C[0..385]
  if (tid == 0) {
    C[0] = 0.f;
    float acc = 0.f;
    for (int m = 0; m < Mlen; ++m) {
      acc += rn[(m * 3 + 2) * 2 + 0] + un[(m * 3 + 2) * 2 + 1];
      C[m + 1] = acc;
    }
    C[Mlen + 1] = 0.f;  // C[385]
  }
  __syncthreads();

  // per-state source/target factors + unnormalized init
  for (int k = tid; k < Kst; k += 256) {
    const int g = (k >= Mlen) ? 1 : 0;
    const int m = g ? (k - Mlen) : k;
    const int s = m + 1 - g;
    // lre/lue with boundary row 384: pad_r=[NEG,0], pad_u=[0,NEG]
    auto LRE = [&](int idx, int j, int x) -> float {
      return (idx < Mlen) ? rn[(idx * 3 + j) * 2 + x] : (x ? 0.f : NEGL);
    };
    auto LUE = [&](int idx, int j, int x) -> float {
      return (idx < Mlen) ? un[(idx * 3 + j) * 2 + x] : (x ? NEGL : 0.f);
    };
    const float sstay = LRE(s, g, 0), sdel = LUE(s, g, 1);
    srcA[k] = LRE(s, g, 1);                  // direct -> insert target
    srcB[k] = sstay + LUE(s, g, 0);          // direct -> match target
    rowb[k] = sstay + sdel - C[s + 1];       // chain base
    tt[k]   = g ? LRE(m, 2, 1) : (LRE(m, 2, 0) + LUE(m, 2, 0));
    float iv;
    if (m == 0) iv = g ? rn[1] : (rn[0] + un[0]);
    else        iv = rn[0] + un[1] - C[1] + C[m] + tt[k];
    initv[k] = iv;
  }
  __syncthreads();

  // normalize init (block logsumexp over 769)
  float lmax = -3.0e38f;
  for (int k = tid; k < Kst; k += 256) lmax = fmaxf(lmax, initv[k]);
  red[tid] = lmax; __syncthreads();
  for (int s2 = 128; s2 > 0; s2 >>= 1) {
    if (tid < s2) red[tid] = fmaxf(red[tid], red[tid + s2]);
    __syncthreads();
  }
  const float imax = red[0]; __syncthreads();
  float lsum = 0.f;
  for (int k = tid; k < Kst; k += 256) lsum += expf(initv[k] - imax);
  red[tid] = lsum; __syncthreads();
  for (int s2 = 128; s2 > 0; s2 >>= 1) {
    if (tid < s2) red[tid] += red[tid + s2];
    __syncthreads();
  }
  const float ilse = imax + logf(red[0]); __syncthreads();
  for (int k = tid; k < Kst; k += 256) initv[k] -= ilse;

  // normalized emissions, transposed [a][kcol]
  for (int k = tid; k < Kst; k += 256) {
    const float* src = (k < Mlen) ? (pre + k * Aal) : (iseq + (k - Mlen) * Aal);
    float mx = -3.0e38f;
    for (int a = 0; a < Aal; ++a) mx = fmaxf(mx, src[a]);
    float s = 0.f;
    for (int a = 0; a < Aal; ++a) s += expf(src[a] - mx);
    const float l = mx + logf(s);
    for (int a = 0; a < Aal; ++a) obs[a * KCOL + k] = src[a] - l;
  }
}

// ---------------- kernel 2: one-hot -> letter ids ----------------
__global__ void hmm_letters_kernel(const float* __restrict__ seq,
                                   char* __restrict__ ws) {
  int* let = (int*)(ws + OFF_LET);
  const int b = blockIdx.x, t = threadIdx.x;
  const float* row = seq + ((size_t)b * Ll + t) * Aal;
  int lt = -1;
  for (int a = 0; a < Aal; ++a)
    if (row[a] > 0.5f) lt = a;
  let[b * Ll + t] = lt;
}

// ---------------- kernel 3: transition row -> packed f16 B tiles ----------------
__global__ void hmm_trans_kernel(char* __restrict__ ws) {
  const float* srcA = (const float*)(ws + OFF_SRCA);
  const float* srcB = (const float*)(ws + OFF_SRCB);
  const float* rowb = (const float*)(ws + OFF_ROWB);
  const float* tt   = (const float*)(ws + OFF_TT);
  const float* C    = (const float*)(ws + OFF_C);
  _Float16* Tp      = (_Float16*)(ws + OFF_TPACK);
  const int i = blockIdx.x;                // source state row
  const int tid = threadIdx.x;
  const int gi = (i >= Mlen) ? 1 : 0;
  const int mi = gi ? (i - Mlen) : i;
  const int si = mi + 1 - gi;
  const float sA = srcA[i], sB = srcB[i], rb = rowb[i];
  __shared__ float red[256];

  float lg[4];
#pragma unroll
  for (int jj = 0; jj < 4; ++jj) {
    const int j = tid + jj * 256;
    float v = NEGL;
    if (j < Kst) {
      const int gj = (j >= Mlen) ? 1 : 0;
      const int mj = gj ? (j - Mlen) : j;
      if (mj == si)      v = gj ? sA : sB;
      else if (mj > si)  v = rb + C[mj] + tt[j];
    }
    lg[jj] = v;
  }
  float lmax = fmaxf(fmaxf(lg[0], lg[1]), fmaxf(lg[2], lg[3]));
  red[tid] = lmax; __syncthreads();
  for (int s2 = 128; s2 > 0; s2 >>= 1) {
    if (tid < s2) red[tid] = fmaxf(red[tid], red[tid + s2]);
    __syncthreads();
  }
  const float rmax = red[0]; __syncthreads();
  float lsum = 0.f;
#pragma unroll
  for (int jj = 0; jj < 4; ++jj) lsum += expf(lg[jj] - rmax);
  red[tid] = lsum; __syncthreads();
  for (int s2 = 128; s2 > 0; s2 >>= 1) {
    if (tid < s2) red[tid] += red[tid + s2];
    __syncthreads();
  }
  const float lse = rmax + logf(red[0]);

  // write exp(normalized) into WMMA-B fragment layout:
  // tile = j/16, kt = i/32; lane = (j%16) + 16*((i%32)/16); elem = i%16
  const int kt = i >> 5, kk = i & 31, elem = kk & 15, lhalf = (kk >> 4) << 4;
#pragma unroll
  for (int jj = 0; jj < 4; ++jj) {
    const int j = tid + jj * 256;
    if (j < Kst) {
      const float pv = expf(lg[jj] - lse);
      const int tile = j >> 4;
      const int lane = (j & 15) + lhalf;
      Tp[((size_t)(tile * NKT + kt)) * 512 + lane * 16 + elem] = (_Float16)pv;
    }
  }
}

// ---------------- kernel 4: forward scan (WMMA) ----------------
__global__ void __launch_bounds__(512, 4)
hmm_scan_kernel(const char* __restrict__ ws,
                const float* __restrict__ local_scale,
                float* __restrict__ out) {
  extern __shared__ char smem[];
  float*    alpha = (float*)(smem + LDS_ALPHA);   // [16][784]
  char*     eA    = smem + LDS_EA;                // A fragments, NKT*1024 B
  float*    mxs   = (float*)(smem + LDS_MX);      // [16]
  int*      lts   = (int*)(smem + LDS_LT);        // [16]

  const _Float16* Tp  = (const _Float16*)(ws + OFF_TPACK);
  const float* obs    = (const float*)(ws + OFF_OBS);
  const float* initv  = (const float*)(ws + OFF_INIT);
  const int*   let    = (const int*)(ws + OFF_LET);

  const int tid  = threadIdx.x;
  const int w    = tid >> 5;       // wave id 0..15 (= batch row within group)
  const int lane = tid & 31;
  const int g16  = blockIdx.x * 16;

  // this wave's three always-valid N tiles (w, w+16, w+32 are all < 49)
  const int t0 = w, t1 = w + 16, t2 = w + 32;
  const bool has48 = (w == 0);     // tile 48 (only real column: state 768)

  // init: alpha = init + emission(t=0); pad cols -> -1e30
  for (int idx = tid; idx < 16 * KCOL; idx += 512) {
    const int row = idx / KCOL, col = idx - row * KCOL;
    float v = -1.0e30f;
    if (col < Kst) {
      const int lt = let[(g16 + row) * Ll + 0];
      const float e = (lt >= 0) ? obs[lt * KCOL + col] : 0.f;
      v = initv[col] + e;
    }
    alpha[row * KCOL + col] = v;
  }
  __syncthreads();

  for (int t = 1; t < Ll; ++t) {
    // ---- phase 1: wave w owns batch row w: max, exp, pack A fragments ----
    float m = -3.0e38f;
    for (int k = lane; k < KCOL; k += 32) m = fmaxf(m, alpha[w * KCOL + k]);
#pragma unroll
    for (int off = 16; off > 0; off >>= 1) m = fmaxf(m, __shfl_xor(m, off, 32));
    if (lane == 0) { mxs[w] = m; lts[w] = let[(g16 + w) * Ll + t]; }
    for (int k = lane; k < KROW; k += 32) {
      float e = 0.f;
      if (k < Kst) e = __expf(alpha[w * KCOL + k] - m);
      const int kt = k >> 5, kk = k & 31, sub = kk & 15, half = kk >> 4;
      const int laneA = w + (half << 4);
      const int elem  = sub;
      ((_Float16*)eA)[kt * 512 + laneA * 16 + elem] = (_Float16)e;
    }
    __syncthreads();

    // ---- phase 2: GEMM (16 x 800) @ (800 x 784) over this wave's N tiles ----
    // The B base addresses are t-invariant. Launder a zero byte-offset
    // through an SGPR so (a) LICM cannot hoist the loads out of the t loop
    // (round-2 scratch-spill bug), while (b) pointer provenance is kept so
    // the loads stay GLOBAL (round-3 emitted flat_load, polluting DScnt).
    unsigned long long zofs = 0;
    asm volatile("" : "+s"(zofs));

    const _Float16* B0 = Tp + (size_t)t0 * NKT * 512 + lane * 16 + zofs;
    const _Float16* B1 = Tp + (size_t)t1 * NKT * 512 + lane * 16 + zofs;
    const _Float16* B2 = Tp + (size_t)t2 * NKT * 512 + lane * 16 + zofs;

    const v8f vzero = {0.f, 0.f, 0.f, 0.f, 0.f, 0.f, 0.f, 0.f};
    v8f acc0 = vzero, acc1 = vzero, acc2 = vzero;

    for (int kt = 0; kt < NKT; ++kt) {
      // A fragment from LDS
      Frag a;
      const float4* ap = (const float4*)(eA + kt * 1024 + lane * 32);
      a.f[0] = ap[0]; a.f[1] = ap[1];

      // B fragments into DISTINCT registers so loads overlap with wmma
      Frag b0, b1, b2;
      const float4* bp0 = (const float4*)(B0 + (size_t)kt * 512);
      const float4* bp1 = (const float4*)(B1 + (size_t)kt * 512);
      const float4* bp2 = (const float4*)(B2 + (size_t)kt * 512);
      b0.f[0] = bp0[0]; b0.f[1] = bp0[1];
      b1.f[0] = bp1[0]; b1.f[1] = bp1[1];
      b2.f[0] = bp2[0]; b2.f[1] = bp2[1];

      acc0 = __builtin_amdgcn_wmma_f32_16x16x32_f16(false, a.h, false, b0.h,
                                                    (short)0, acc0, false, false);
      acc1 = __builtin_amdgcn_wmma_f32_16x16x32_f16(false, a.h, false, b1.h,
                                                    (short)0, acc1, false, false);
      acc2 = __builtin_amdgcn_wmma_f32_16x16x32_f16(false, a.h, false, b2.h,
                                                    (short)0, acc2, false, false);
    }

    // ---- write back: alpha_new = log(max(pr,1e-30)) + mx + emission ----
    {
      const int colb = (lane & 15);
      const int row0 = (lane >> 4) * 8;
#pragma unroll
      for (int r = 0; r < 8; ++r) {
        const int row = row0 + r;
        const int lt = lts[row];
        const float mxr = mxs[row];
        int col = t0 * 16 + colb;
        float e = (lt >= 0) ? obs[lt * KCOL + col] : 0.f;
        alpha[row * KCOL + col] = __logf(fmaxf(acc0[r], 1.0e-30f)) + mxr + e;
        col = t1 * 16 + colb;
        e = (lt >= 0) ? obs[lt * KCOL + col] : 0.f;
        alpha[row * KCOL + col] = __logf(fmaxf(acc1[r], 1.0e-30f)) + mxr + e;
        col = t2 * 16 + colb;
        e = (lt >= 0) ? obs[lt * KCOL + col] : 0.f;
        alpha[row * KCOL + col] = __logf(fmaxf(acc2[r], 1.0e-30f)) + mxr + e;
      }
    }

    // ---- wave 0 only: tile 48 (one real column, state 768) as a mini-pass,
    // reusing the (still valid) A fragments; keeps 16 VGPRs out of the hot loop.
    if (has48) {
      const _Float16* B3 = Tp + (size_t)48 * NKT * 512 + lane * 16 + zofs;
      v8f acc3 = vzero;
      for (int kt = 0; kt < NKT; ++kt) {
        Frag a;
        const float4* ap = (const float4*)(eA + kt * 1024 + lane * 32);
        a.f[0] = ap[0]; a.f[1] = ap[1];
        Frag b3;
        const float4* bp3 = (const float4*)(B3 + (size_t)kt * 512);
        b3.f[0] = bp3[0]; b3.f[1] = bp3[1];
        acc3 = __builtin_amdgcn_wmma_f32_16x16x32_f16(false, a.h, false, b3.h,
                                                      (short)0, acc3, false, false);
      }
      const int col = 48 * 16 + (lane & 15);
      if (col < Kst) {
        const int row0 = (lane >> 4) * 8;
#pragma unroll
        for (int r = 0; r < 8; ++r) {
          const int row = row0 + r;
          const int lt = lts[row];
          const float e = (lt >= 0) ? obs[lt * KCOL + col] : 0.f;
          alpha[row * KCOL + col] =
              __logf(fmaxf(acc3[r], 1.0e-30f)) + mxs[row] + e;
        }
      }
    }
    __syncthreads();
  }

  // ---- final logsumexp per batch row ----
  {
    float m = -3.0e38f;
    for (int k = lane; k < Kst; k += 32) m = fmaxf(m, alpha[w * KCOL + k]);
#pragma unroll
    for (int off = 16; off > 0; off >>= 1) m = fmaxf(m, __shfl_xor(m, off, 32));
    float s = 0.f;
    for (int k = lane; k < Kst; k += 32) s += __expf(alpha[w * KCOL + k] - m);
#pragma unroll
    for (int off = 16; off > 0; off >>= 1) s += __shfl_xor(s, off, 32);
    if (lane == 0) out[g16 + w] = local_scale[0] * (m + logf(s));
  }
}

// ---------------- launch ----------------
extern "C" void kernel_launch(void* const* d_in, const int* in_sizes, int n_in,
                              void* d_out, int out_size, void* d_ws, size_t ws_size,
                              hipStream_t stream) {
  const float* pre   = (const float*)d_in[0];   // (384,21)
  const float* iseq  = (const float*)d_in[1];   // (385,21)
  const float* ins   = (const float*)d_in[2];   // (384,3,2)
  const float* del   = (const float*)d_in[3];   // (384,3,2)
  const float* seq   = (const float*)d_in[4];   // (64,256,21)
  const float* scale = (const float*)d_in[5];   // (1,)
  float* out = (float*)d_out;
  char*  ws  = (char*)d_ws;

  hmm_setup_kernel<<<1, 256, 0, stream>>>(pre, iseq, ins, del, ws);
  hmm_letters_kernel<<<Bb, Ll, 0, stream>>>(seq, ws);
  hipMemsetAsync(ws + OFF_TPACK, 0, SZ_TPACK, stream);
  hmm_trans_kernel<<<Kst, 256, 0, stream>>>(ws);
  hmm_scan_kernel<<<Bb / 16, 512, LDS_BYTES, stream>>>(ws, scale, out);
}
